// MultiHeadAttention_41420664602857
// MI455X (gfx1250) — matrix operand order repaired and checked by reading, not compiled, not run on hardware
//
#include <hip/hip_runtime.h>

// MHA forward for MI455X (gfx1250): bf16 WMMA everywhere, flash-attention
// structure, fp32 accumulation. LDS-staged (async-to-LDS, double buffered)
// weight tiles shared by all 8 waves of a block, 16x64 per-wave strips
// sharing one A-fragment across 4 WMMAs, reuse_a hints.
//
// Dims: B=4, S=2048, E=1024, H=16, D=64, M=B*S=8192.

typedef __bf16 bf16;
typedef __attribute__((ext_vector_type(16))) __bf16 v16bf;
typedef __attribute__((ext_vector_type(8)))  __bf16 v8bf;
typedef __attribute__((ext_vector_type(8)))  float  v8f;
typedef __attribute__((ext_vector_type(4)))  int    v4i;

#define NB   4
#define SEQ  2048
#define EMB  1024
#define NH   16
#define HD   64
#define MROWS (NB * SEQ)   // 8192

#if __has_builtin(__builtin_amdgcn_global_load_async_to_lds_b128) && \
    __has_builtin(__builtin_amdgcn_s_wait_asynccnt)
#define HAS_ASYNC_LDS 1
#endif

union FragU { v16bf v; v8bf h[2]; };

__device__ __forceinline__ v8f zero8() {
  v8f z;
#pragma unroll
  for (int i = 0; i < 8; ++i) z[i] = 0.0f;
  return z;
}

__device__ __forceinline__ v8f wmma_bf16(v16bf a, v16bf b, v8f c) {
  return __builtin_amdgcn_wmma_f32_16x16x32_bf16(
      false, a, false, b, (short)0, c, /*reuse_a=*/false, /*reuse_b=*/false);
}
// Variant with the VOP3P "reuse A" hint: legal when the NEXT instruction is
// an identical WMMA consuming the same A operand.
__device__ __forceinline__ v8f wmma_bf16_ra(v16bf a, v16bf b, v8f c) {
  return __builtin_amdgcn_wmma_f32_16x16x32_bf16(
      false, a, false, b, (short)0, c, /*reuse_a=*/true, /*reuse_b=*/false);
}

// A-fragment (16x32, MxK) from row-major [rows, ld] bf16.
// Lane l: m=l&15, kh=l>>4; elems 0..7 -> K=k0+kh*8+j, elems 8..15 -> K=k0+16+kh*8+j.
__device__ __forceinline__ v16bf load_frag_a(const bf16* p, int ld, int row0, int k0) {
  int lane = threadIdx.x & 31;
  int m  = lane & 15;
  int kh = lane >> 4;
  const bf16* q = p + (size_t)(row0 + m) * ld + k0 + kh * 8;
  FragU u;
  u.h[0] = *(const v8bf*)q;
  u.h[1] = *(const v8bf*)(q + 16);
  return u.v;
}

// B-fragment (32x16, KxN) from TRANSPOSED matrix B^T stored row-major [N, ld].
// Lane l: n=l&15, kh=l>>4; elems 0..7 -> K=k0+kh*8+j, elems 8..15 -> K=k0+16+kh*8+j.
__device__ __forceinline__ v16bf load_frag_bt(const bf16* p, int ld, int n0, int k0) {
  int lane = threadIdx.x & 31;
  int n  = lane & 15;
  int kh = lane >> 4;
  const bf16* q = p + (size_t)(n0 + n) * ld + k0 + kh * 8;
  FragU u;
  u.h[0] = *(const v8bf*)q;
  u.h[1] = *(const v8bf*)(q + 16);
  return u.v;
}

// Cooperative 16-byte stage of one chunk into LDS. Async-to-LDS path on
// gfx1250 (tracked by ASYNCcnt, no VGPR round trip), sync fallback otherwise.
__device__ __forceinline__ void stage_b128(const bf16* g, bf16* l) {
#if defined(HAS_ASYNC_LDS)
  __builtin_amdgcn_global_load_async_to_lds_b128(
      (v4i*)g, (v4i*)l, /*offset=*/0, /*cpol=*/0);
#else
  *(v8bf*)l = *(const v8bf*)g;
#endif
}
__device__ __forceinline__ void stage_commit() {
#if defined(HAS_ASYNC_LDS)
  __builtin_amdgcn_s_wait_asynccnt(0);
#endif
  __syncthreads();
}

// ---------------- conversion / layout kernels ----------------

__global__ void cvt_f32_bf16(const float* __restrict__ src, bf16* __restrict__ dst, int n) {
  int i = blockIdx.x * blockDim.x + threadIdx.x;
  int stride = gridDim.x * blockDim.x;
  for (; i < n; i += stride) dst[i] = (bf16)src[i];
}

// W: [H, E, D] f32 -> Wt: [H, D, E] bf16  (B^T layout per head)
__global__ void cvt_w_hed_hde(const float* __restrict__ w, bf16* __restrict__ wt) {
  int i = blockIdx.x * blockDim.x + threadIdx.x;
  int stride = gridDim.x * blockDim.x;
  const int n = NH * EMB * HD;
  for (; i < n; i += stride) {
    int h = i / (EMB * HD);
    int r = i % (EMB * HD);
    int e = r / HD;
    int d = r % HD;
    wt[((size_t)h * HD + d) * EMB + e] = (bf16)w[i];
  }
}

// Wo: [E(in), E(out)] f32 -> Wot: [E(out), E(in)] bf16
__global__ void cvt_wo_t(const float* __restrict__ w, bf16* __restrict__ wt) {
  int i = blockIdx.x * blockDim.x + threadIdx.x;
  int stride = gridDim.x * blockDim.x;
  const int n = EMB * EMB;
  for (; i < n; i += stride) {
    int k = i / EMB;
    int nn = i % EMB;
    wt[(size_t)nn * EMB + k] = (bf16)w[i];
  }
}

// ---------------- QKV projection GEMM ----------------
// grid: (MROWS/16/8, H, 3), block 256 (8 waves).
// Each wave computes a 16x64 strip (entire head dim) of one (h, q/k/v).
// All 8 waves share the 64x32 weight k-tile, staged in LDS (double buffered,
// async-to-LDS). Q,K stored [B,H,S,D] bf16; V stored transposed [B,H,D,S].
__global__ void qkv_gemm(const bf16* __restrict__ X,
                         const bf16* __restrict__ Wqt, const bf16* __restrict__ Wkt,
                         const bf16* __restrict__ Wvt,
                         const float* __restrict__ bq, const float* __restrict__ bk,
                         const float* __restrict__ bv,
                         bf16* __restrict__ Q, bf16* __restrict__ K,
                         bf16* __restrict__ Vt) {
  __shared__ bf16 bstage[2][HD * 32];   // 2 x 4 KB

  int wave = threadIdx.x >> 5;
  int lane = threadIdx.x & 31;
  int h = blockIdx.y;
  int which = blockIdx.z;               // 0=q 1=k 2=v
  int mt = blockIdx.x * 8 + wave;       // 0..511
  int m0 = mt * 16;

  const bf16* Wt = (which == 0) ? Wqt : (which == 1) ? Wkt : Wvt;
  const float* bias = (which == 0) ? bq : (which == 1) ? bk : bv;
  const bf16* wh = Wt + (size_t)h * HD * EMB;

  // cooperative stage indices: 64 rows x 32 cols bf16 = 256 x 16B chunks
  int tr = threadIdx.x >> 2;            // 0..63 (row of B^T tile)
  int tc = (threadIdx.x & 3) * 8;       // 0,8,16,24 (col element)

  stage_b128(wh + (size_t)tr * EMB + tc, &bstage[0][tr * 32 + tc]);
  stage_commit();

  v8f acc[4];
#pragma unroll
  for (int st = 0; st < 4; ++st) acc[st] = zero8();

  int buf = 0;
  const int KSTEPS = EMB / 32;
  for (int ks = 0; ks < KSTEPS; ++ks) {
    int k0 = ks * 32;
    if (ks + 1 < KSTEPS)
      stage_b128(wh + (size_t)tr * EMB + (k0 + 32) + tc,
                 &bstage[buf ^ 1][tr * 32 + tc]);

    v16bf a = load_frag_a(X, EMB, m0, k0);
    const bf16* bt = bstage[buf];
    acc[0] = wmma_bf16_ra(a, load_frag_bt(bt, 32, 0,  0), acc[0]);
    acc[1] = wmma_bf16_ra(a, load_frag_bt(bt, 32, 16, 0), acc[1]);
    acc[2] = wmma_bf16_ra(a, load_frag_bt(bt, 32, 32, 0), acc[2]);
    acc[3] = wmma_bf16   (a, load_frag_bt(bt, 32, 48, 0), acc[3]);

    stage_commit();
    buf ^= 1;
  }

  int n = lane & 15, mh = lane >> 4;
#pragma unroll
  for (int st = 0; st < 4; ++st) {
    int n0 = st * 16;
    float bval = bias[h * HD + n0 + n];
#pragma unroll
    for (int i = 0; i < 8; ++i) {
      int mrow = m0 + i + 8 * mh;       // global row in [0, MROWS)
      int bb = mrow >> 11;              // / SEQ
      int ss = mrow & (SEQ - 1);
      float val = acc[st][i] + bval;
      if (which == 2) {
        Vt[(((size_t)bb * NH + h) * HD + (n0 + n)) * SEQ + ss] = (bf16)val;
      } else {
        bf16* dst = (which == 0) ? Q : K;
        dst[(((size_t)bb * NH + h) * SEQ + ss) * HD + (n0 + n)] = (bf16)val;
      }
    }
  }
}

// ---------------- flash attention ----------------
// grid: (SEQ/64, H, B), block 128 (4 waves). Each wave owns 16 query rows.
// C-fragment layout: VGPR i of lane l holds (row = i + 8*(l>>4), col = l&15).
__global__ void flash_attn(const bf16* __restrict__ Q, const bf16* __restrict__ K,
                           const bf16* __restrict__ Vt, bf16* __restrict__ O) {
  __shared__ bf16 pbuf[4][16 * 32];     // per-wave P staging (C-layout -> A-layout)

  int wave = threadIdx.x >> 5;
  int lane = threadIdx.x & 31;
  int h = blockIdx.y, b = blockIdx.z;
  int m0 = blockIdx.x * 64 + wave * 16;

  const bf16* qh = Q + (((size_t)b * NH + h) * SEQ) * HD;
  const bf16* kh = K + (((size_t)b * NH + h) * SEQ) * HD;
  const bf16* vh = Vt + (((size_t)b * NH + h) * HD) * SEQ;

  v16bf aq0 = load_frag_a(qh, HD, m0, 0);
  v16bf aq1 = load_frag_a(qh, HD, m0, 32);

  v8f o0 = zero8(), o1 = zero8(), o2 = zero8(), o3 = zero8();
  float mrow[8], lrow[8];
#pragma unroll
  for (int i = 0; i < 8; ++i) { mrow[i] = -1e30f; lrow[i] = 0.0f; }

  bf16* pl = pbuf[wave];
  int n = lane & 15, mh = lane >> 4;

  for (int t0 = 0; t0 < SEQ; t0 += 32) {
    v8f s0 = zero8(), s1 = zero8();
    s0 = wmma_bf16(aq0, load_frag_bt(kh, HD, t0, 0), s0);
    s0 = wmma_bf16(aq1, load_frag_bt(kh, HD, t0, 32), s0);
    s1 = wmma_bf16(aq0, load_frag_bt(kh, HD, t0 + 16, 0), s1);
    s1 = wmma_bf16(aq1, load_frag_bt(kh, HD, t0 + 16, 32), s1);

    float p0[8], p1[8], tmax[8];
#pragma unroll
    for (int i = 0; i < 8; ++i) {
      p0[i] = s0[i] * 0.125f;           // 1/sqrt(64)
      p1[i] = s1[i] * 0.125f;
      tmax[i] = fmaxf(p0[i], p1[i]);
    }
#pragma unroll
    for (int i = 0; i < 8; ++i) {
      tmax[i] = fmaxf(tmax[i], __shfl_xor(tmax[i], 1, 32));
      tmax[i] = fmaxf(tmax[i], __shfl_xor(tmax[i], 2, 32));
      tmax[i] = fmaxf(tmax[i], __shfl_xor(tmax[i], 4, 32));
      tmax[i] = fmaxf(tmax[i], __shfl_xor(tmax[i], 8, 32));
    }
#pragma unroll
    for (int i = 0; i < 8; ++i) {
      float mn = fmaxf(mrow[i], tmax[i]);
      float alpha = __expf(mrow[i] - mn);
      mrow[i] = mn;
      p0[i] = __expf(p0[i] - mn);
      p1[i] = __expf(p1[i] - mn);
      float rs = p0[i] + p1[i];
      rs += __shfl_xor(rs, 1, 32);
      rs += __shfl_xor(rs, 2, 32);
      rs += __shfl_xor(rs, 4, 32);
      rs += __shfl_xor(rs, 8, 32);
      lrow[i] = lrow[i] * alpha + rs;
      o0[i] *= alpha; o1[i] *= alpha; o2[i] *= alpha; o3[i] *= alpha;
    }
#pragma unroll
    for (int i = 0; i < 8; ++i) {
      pl[(i + 8 * mh) * 32 + n]      = (bf16)p0[i];
      pl[(i + 8 * mh) * 32 + 16 + n] = (bf16)p1[i];
    }
    v16bf pa = load_frag_a(pl, 32, 0, 0);
    o0 = wmma_bf16_ra(pa, load_frag_bt(vh, SEQ, 0,  t0), o0);
    o1 = wmma_bf16_ra(pa, load_frag_bt(vh, SEQ, 16, t0), o1);
    o2 = wmma_bf16_ra(pa, load_frag_bt(vh, SEQ, 32, t0), o2);
    o3 = wmma_bf16   (pa, load_frag_bt(vh, SEQ, 48, t0), o3);
  }

  float rinv[8];
#pragma unroll
  for (int i = 0; i < 8; ++i) rinv[i] = 1.0f / lrow[i];

  size_t rbase = (size_t)b * SEQ + m0;
#pragma unroll
  for (int i = 0; i < 8; ++i) {
    bf16* dst = O + (rbase + i + 8 * mh) * EMB + h * HD;
    dst[ 0 + n] = (bf16)(o0[i] * rinv[i]);
    dst[16 + n] = (bf16)(o1[i] * rinv[i]);
    dst[32 + n] = (bf16)(o2[i] * rinv[i]);
    dst[48 + n] = (bf16)(o3[i] * rinv[i]);
  }
}

// ---------------- output projection ----------------
// grid: (MROWS/16/8, EMB/64), block 256 (8 waves). Wave computes 16x64 strip;
// all 8 waves share the 64x32 Wo^T k-tile staged in LDS (double buffered).
__global__ void out_proj(const bf16* __restrict__ Ob, const bf16* __restrict__ Wot,
                         const float* __restrict__ bo, float* __restrict__ out) {
  __shared__ bf16 bstage[2][64 * 32];

  int wave = threadIdx.x >> 5;
  int lane = threadIdx.x & 31;
  int mt = blockIdx.x * 8 + wave;       // 0..511
  int m0 = mt * 16;
  int nb = blockIdx.y * 64;             // n-range base for this block

  const bf16* wbase = Wot + (size_t)nb * EMB;
  int tr = threadIdx.x >> 2;
  int tc = (threadIdx.x & 3) * 8;

  stage_b128(wbase + (size_t)tr * EMB + tc, &bstage[0][tr * 32 + tc]);
  stage_commit();

  v8f acc[4];
#pragma unroll
  for (int st = 0; st < 4; ++st) acc[st] = zero8();

  int buf = 0;
  const int KSTEPS = EMB / 32;
  for (int ks = 0; ks < KSTEPS; ++ks) {
    int k0 = ks * 32;
    if (ks + 1 < KSTEPS)
      stage_b128(wbase + (size_t)tr * EMB + (k0 + 32) + tc,
                 &bstage[buf ^ 1][tr * 32 + tc]);

    v16bf a = load_frag_a(Ob, EMB, m0, k0);
    const bf16* bt = bstage[buf];
    acc[0] = wmma_bf16_ra(a, load_frag_bt(bt, 32, 0,  0), acc[0]);
    acc[1] = wmma_bf16_ra(a, load_frag_bt(bt, 32, 16, 0), acc[1]);
    acc[2] = wmma_bf16_ra(a, load_frag_bt(bt, 32, 32, 0), acc[2]);
    acc[3] = wmma_bf16   (a, load_frag_bt(bt, 32, 48, 0), acc[3]);

    stage_commit();
    buf ^= 1;
  }

  int n = lane & 15, mh = lane >> 4;
#pragma unroll
  for (int st = 0; st < 4; ++st) {
    int n0 = nb + st * 16;
    float bval = bo[n0 + n];
#pragma unroll
    for (int i = 0; i < 8; ++i)
      out[(size_t)(m0 + i + 8 * mh) * EMB + n0 + n] = acc[st][i] + bval;
  }
}

// ---------------- host launch ----------------
extern "C" void kernel_launch(void* const* d_in, const int* in_sizes, int n_in,
                              void* d_out, int out_size, void* d_ws, size_t ws_size,
                              hipStream_t stream) {
  const float* X  = (const float*)d_in[0];
  const float* Wq = (const float*)d_in[1];
  const float* bq = (const float*)d_in[2];
  const float* Wk = (const float*)d_in[3];
  const float* bk = (const float*)d_in[4];
  const float* Wv = (const float*)d_in[5];
  const float* bv = (const float*)d_in[6];
  const float* Wo = (const float*)d_in[7];
  const float* bo = (const float*)d_in[8];
  float* out = (float*)d_out;

  char* ws = (char*)d_ws;
  const size_t nXE = (size_t)MROWS * EMB;
  const size_t nWH = (size_t)NH * HD * EMB;
  const size_t nWO = (size_t)EMB * EMB;
  const size_t nQ  = (size_t)NB * NH * SEQ * HD;

  bf16* Xb  = (bf16*)ws;  ws += nXE * sizeof(bf16);
  bf16* Wqt = (bf16*)ws;  ws += nWH * sizeof(bf16);
  bf16* Wkt = (bf16*)ws;  ws += nWH * sizeof(bf16);
  bf16* Wvt = (bf16*)ws;  ws += nWH * sizeof(bf16);
  bf16* Wot = (bf16*)ws;  ws += nWO * sizeof(bf16);
  bf16* Qb  = (bf16*)ws;  ws += nQ  * sizeof(bf16);
  bf16* Kb  = (bf16*)ws;  ws += nQ  * sizeof(bf16);
  bf16* Vtb = (bf16*)ws;  ws += nQ  * sizeof(bf16);
  bf16* Ob  = (bf16*)ws;  ws += nXE * sizeof(bf16);

  cvt_f32_bf16<<<2048, 256, 0, stream>>>(X, Xb, (int)nXE);
  cvt_w_hed_hde<<<1024, 256, 0, stream>>>(Wq, Wqt);
  cvt_w_hed_hde<<<1024, 256, 0, stream>>>(Wk, Wkt);
  cvt_w_hed_hde<<<1024, 256, 0, stream>>>(Wv, Wvt);
  cvt_wo_t<<<1024, 256, 0, stream>>>(Wo, Wot);

  qkv_gemm<<<dim3(MROWS / 16 / 8, NH, 3), 256, 0, stream>>>(
      Xb, Wqt, Wkt, Wvt, bq, bk, bv, Qb, Kb, Vtb);

  flash_attn<<<dim3(SEQ / 64, NH, NB), 128, 0, stream>>>(Qb, Kb, Vtb, Ob);

  out_proj<<<dim3(MROWS / 16 / 8, EMB / 64), 256, 0, stream>>>(Ob, Wot, bo, out);
}
